// GCN_trad_11871289606368
// MI455X (gfx1250) — matrix operand order repaired and checked by reading, not compiled, not run on hardware
//
#include <hip/hip_runtime.h>
#include <hip/hip_bf16.h>

typedef __attribute__((ext_vector_type(16))) __bf16 v16bf;
typedef __attribute__((ext_vector_type(8)))  __bf16 v8bf;
typedef __attribute__((ext_vector_type(8)))  float  v8f;

// ---------------------------------------------------------------------------
// split a f32 into bf16 hi + bf16 lo  (bf16x3 fp32-emulation building block)
// ---------------------------------------------------------------------------
__device__ __forceinline__ void bsplit(float x, __bf16& hi, __bf16& lo) {
    __bf16 h = (__bf16)x;
    hi = h;
    lo = (__bf16)(x - (float)h);
}

// elementwise f32 -> (bf16 hi, bf16 lo); X stays row-major so K is contiguous
__global__ void split_k(const float* __restrict__ x, __bf16* __restrict__ hi,
                        __bf16* __restrict__ lo, int n) {
    int i = blockIdx.x * blockDim.x + threadIdx.x;
    if (i >= n) return;
    bsplit(x[i], hi[i], lo[i]);
}

// Pre-pack W[Fin,Fout] into per-lane WMMA B-operand order (contiguous k, same order as A):
//   idx = ((colTile*numChunks + chunk)*32 + lane)*16 + e
//   e = 2j+p  ->  K = chunk*32 + 2j + 16*(lane>>4) + p,  col = colTile*16 + (lane&15)
// so each lane's 16 bf16 B elements are 32 contiguous bytes (two b128 loads).
__global__ void pack_w_k(const float* __restrict__ W, __bf16* __restrict__ whi,
                         __bf16* __restrict__ wlo, int Fin, int Fout) {
    int idx = blockIdx.x * blockDim.x + threadIdx.x;
    if (idx >= Fin * Fout) return;
    int e    = idx & 15;
    int lane = (idx >> 4) & 31;
    int rest = idx >> 9;                 // colTile*numChunks + chunk
    int numChunks = Fin >> 5;
    int chunk   = rest % numChunks;
    int colTile = rest / numChunks;
    int k = chunk * 32 + (e & ~1) + 16 * (lane >> 4) + (e & 1);
    int c = (colTile << 4) + (lane & 15);
    __bf16 h, l;
    bsplit(W[(size_t)k * Fout + c], h, l);
    whi[idx] = h;
    wlo[idx] = l;
}

// ---------------------------------------------------------------------------
// Y[N,Fout] = X[N,Fin] @ W[Fin,Fout], bf16x3 via v_wmma_f32_16x16x32_bf16.
// One wave32 per 16x16 tile. A loaded as 4x b128 from split X; B as 4x b128
// from pre-packed W. Requires N % 16 == 0, Fin % 32 == 0, Fout % 16 == 0.
// Dual-output mode (Y2 != null): Y = raw X@W, Y2 = dinv[row]^2 * (X@W) + bias
// (the GCN self-loop term). Otherwise: Y = X@W + bias (+relu).
// ---------------------------------------------------------------------------
__global__ __launch_bounds__(128)
void gemm_wmma(const __bf16* __restrict__ Xhi, const __bf16* __restrict__ Xlo,
               const __bf16* __restrict__ Bph, const __bf16* __restrict__ Bpl,
               const float* __restrict__ bias, float* __restrict__ Y,
               const float* __restrict__ dinv, float* __restrict__ Y2,
               int N, int Fin, int Fout, int doRelu)
{
    const int lane = threadIdx.x & 31;
    const int wave = threadIdx.x >> 5;
    const int gid  = blockIdx.x * (blockDim.x >> 5) + wave;
    const int colTiles = Fout >> 4;
    const int rowTiles = N >> 4;
    if (gid >= rowTiles * colTiles) return;   // wave-uniform exit

    const int row0 = (gid / colTiles) << 4;
    const int col0 = (gid % colTiles) << 4;
    const int m    = lane & 15;
    const int half = lane >> 4;
    const int numChunks = Fin >> 5;

    const __bf16* xh = Xhi + (size_t)(row0 + m) * Fin;
    const __bf16* xl = Xlo + (size_t)(row0 + m) * Fin;
    const __bf16* bph = Bph + ((size_t)(col0 >> 4) * numChunks * 32 + lane) * 16;
    const __bf16* bpl = Bpl + ((size_t)(col0 >> 4) * numChunks * 32 + lane) * 16;

    v8f acc = {};
    for (int c = 0; c < numChunks; ++c) {
        const int k0 = (c << 5) + 16 * half;      // A: lanes0-15 K 0..15
        const int k1 = k0 + 8;                    //    lanes16-31 K 16..31
        v8bf a0h = *(const v8bf*)(xh + k0);
        v8bf a1h = *(const v8bf*)(xh + k1);
        v8bf a0l = *(const v8bf*)(xl + k0);
        v8bf a1l = *(const v8bf*)(xl + k1);
        const __bf16* bh_p = bph + (size_t)c * 512;   // 32 lanes * 16 elems
        const __bf16* bl_p = bpl + (size_t)c * 512;
        v8bf b0h = *(const v8bf*)(bh_p);
        v8bf b1h = *(const v8bf*)(bh_p + 8);
        v8bf b0l = *(const v8bf*)(bl_p);
        v8bf b1l = *(const v8bf*)(bl_p + 8);

        v16bf ah = __builtin_shufflevector(a0h, a1h, 0,1,2,3,4,5,6,7,8,9,10,11,12,13,14,15);
        v16bf al = __builtin_shufflevector(a0l, a1l, 0,1,2,3,4,5,6,7,8,9,10,11,12,13,14,15);
        v16bf bh = __builtin_shufflevector(b0h, b1h, 0,1,2,3,4,5,6,7,8,9,10,11,12,13,14,15);
        v16bf bl = __builtin_shufflevector(b0l, b1l, 0,1,2,3,4,5,6,7,8,9,10,11,12,13,14,15);

        // bf16x3: Ah*Bh + Al*Bh + Ah*Bl (drop Al*Bl, ~2^-22 rel. error)
        acc = __builtin_amdgcn_wmma_f32_16x16x32_bf16(false, ah, false, bh, (short)0, acc, false, false);
        acc = __builtin_amdgcn_wmma_f32_16x16x32_bf16(false, al, false, bh, (short)0, acc, false, false);
        acc = __builtin_amdgcn_wmma_f32_16x16x32_bf16(false, ah, false, bl, (short)0, acc, false, false);
    }

    const float badd = bias ? bias[col0 + m] : 0.0f;
    if (Y2) {
        // GCN layer mode: raw xw for edge-scatter + self-loop-initialized hout
#pragma unroll
        for (int r = 0; r < 8; ++r) {
            int row = row0 + r + 8 * half;
            size_t o = (size_t)row * Fout + col0 + m;
            float v = acc[r];
            float dv = dinv[row];
            Y[o]  = v;
            Y2[o] = dv * dv * v + badd;
        }
    } else {
#pragma unroll
        for (int r = 0; r < 8; ++r) {
            int row = row0 + r + 8 * half;
            float v = acc[r] + badd;
            if (doRelu) v = fmaxf(v, 0.0f);
            Y[(size_t)row * Fout + col0 + m] = v;
        }
    }
}

// ---------------------------------------------------------------------------
// elementwise / sparse helper kernels
// ---------------------------------------------------------------------------
__global__ void fill_k(float* p, float v, int n) {
    int i = blockIdx.x * blockDim.x + threadIdx.x;
    if (i < n) p[i] = v;
}

__global__ void deg_accum_k(const int* __restrict__ dst, const float* __restrict__ w,
                            float* __restrict__ deg, int E) {
    int e = blockIdx.x * blockDim.x + threadIdx.x;
    if (e < E) atomicAdd(&deg[dst[e]], w[e]);
}

__global__ void rsqrt_k(float* p, int n) {
    int i = blockIdx.x * blockDim.x + threadIdx.x;
    if (i < n) p[i] = rsqrtf(p[i]);
}

__global__ void edge_norm_k(const int* __restrict__ src, const int* __restrict__ dst,
                            const float* __restrict__ w, const float* __restrict__ dinv,
                            float* __restrict__ norm, int E) {
    int e = blockIdx.x * blockDim.x + threadIdx.x;
    if (e < E) norm[e] = w[e] * dinv[src[e]] * dinv[dst[e]];
}

// one wave per edge: out[dst] += norm[e] * xw[src]   (lane-coalesced atomics)
__global__ void scatter_k(const int* __restrict__ src, const int* __restrict__ dst,
                          const float* __restrict__ norm, const float* __restrict__ xw,
                          float* __restrict__ out, int E, int F) {
    int t = blockIdx.x * blockDim.x + threadIdx.x;
    int e = t >> 5, lane = t & 31;
    if (e >= E) return;
    int s = src[e], d = dst[e];
    float nm = norm[e];
    const float* xs = xw + (size_t)s * F;
    float* od = out + (size_t)d * F;
    for (int f = lane; f < F; f += 32) atomicAdd(&od[f], nm * xs[f]);
}

__global__ void relu_k(float* p, int n) {
    int i = blockIdx.x * blockDim.x + threadIdx.x;
    if (i < n) p[i] = fmaxf(p[i], 0.0f);
}

// one wave per node: pool[batch[n]] += h[n]; cnt[batch[n]] += 1
__global__ void pool_accum_k(const float* __restrict__ h, const int* __restrict__ batch,
                             float* __restrict__ pool, float* __restrict__ cnt,
                             int N, int F) {
    int t = blockIdx.x * blockDim.x + threadIdx.x;
    int n = t >> 5, lane = t & 31;
    if (n >= N) return;
    int g = batch[n];
    const float* hr = h + (size_t)n * F;
    float* pg = pool + (size_t)g * F;
    for (int f = lane; f < F; f += 32) atomicAdd(&pg[f], hr[f]);
    if (lane == 0) atomicAdd(&cnt[g], 1.0f);
}

__global__ void pool_div_k(float* pool, const float* __restrict__ cnt, int G, int F) {
    int i = blockIdx.x * blockDim.x + threadIdx.x;
    if (i >= G * F) return;
    pool[i] /= fmaxf(cnt[i / F], 1.0f);
}

// eval-mode BatchNorm (running stats mean=0, var=1) + relu
__global__ void bn_relu_k(const float* __restrict__ x, const float* __restrict__ gamma,
                          const float* __restrict__ beta, float* __restrict__ y,
                          int B, int F) {
    int i = blockIdx.x * blockDim.x + threadIdx.x;
    if (i >= B * F) return;
    int f = i % F;
    float s = rsqrtf(1.0f + 1e-5f);
    y[i] = fmaxf(x[i] * s * gamma[f] + beta[f], 0.0f);
}

// tiny [B,F] @ [F,1] + b
__global__ void out_head_k(const float* __restrict__ h, const float* __restrict__ Wo,
                           const float* __restrict__ bo, float* __restrict__ out,
                           int B, int F) {
    int r = blockIdx.x * blockDim.x + threadIdx.x;
    if (r >= B) return;
    float acc = bo[0];
    const float* hr = h + (size_t)r * F;
    for (int f = 0; f < F; ++f) acc += hr[f] * Wo[f];
    out[r] = acc;
}

// ---------------------------------------------------------------------------
// host-side orchestration
// ---------------------------------------------------------------------------
static inline int cdiv(long long a, long long b) { return (int)((a + b - 1) / b); }

struct Scratch {
    __bf16 *Xhi, *Xlo, *Wph, *Wpl;
};

// one dense layer GEMM with split+pack staging
static void gemm_full(const float* X, const float* W, const float* bias,
                      float* Y, const float* dinv, float* Y2,
                      int N, int Fin, int Fout, int doRelu,
                      const Scratch& s, hipStream_t stream)
{
    split_k<<<cdiv((long long)N * Fin, 256), 256, 0, stream>>>(X, s.Xhi, s.Xlo, N * Fin);
    pack_w_k<<<cdiv((long long)Fin * Fout, 256), 256, 0, stream>>>(W, s.Wph, s.Wpl, Fin, Fout);
    int tiles = (N >> 4) * (Fout >> 4);
    gemm_wmma<<<cdiv(tiles, 4), 128, 0, stream>>>(s.Xhi, s.Xlo, s.Wph, s.Wpl,
                                                  bias, Y, dinv, Y2, N, Fin, Fout, doRelu);
}

static void gcn_layer(const float* hin, int Fin, const float* Wt, const float* bias,
                      const int* src, const int* dst, const float* norm, const float* dinv,
                      float* xw, float* hout, int N, int E,
                      const Scratch& s, hipStream_t stream)
{
    const int H = 128;
    // xw = hin @ W ; hout = dinv^2*xw + bias  (fused epilogue)
    gemm_full(hin, Wt, bias, xw, dinv, hout, N, Fin, H, 0, s, stream);
    scatter_k<<<cdiv((long long)E * 32, 256), 256, 0, stream>>>(src, dst, norm, xw, hout, E, H);
    relu_k<<<cdiv((long long)N * H, 256), 256, 0, stream>>>(hout, N * H);
}

extern "C" void kernel_launch(void* const* d_in, const int* in_sizes, int n_in,
                              void* d_out, int out_size, void* d_ws, size_t ws_size,
                              hipStream_t stream)
{
    const float* x_l  = (const float*)d_in[0];
    const int*   ei_l = (const int*)  d_in[1];
    const float* w_l  = (const float*)d_in[2];
    const float* x_s  = (const float*)d_in[3];
    const int*   ei_s = (const int*)  d_in[4];
    const float* w_s  = (const float*)d_in[5];
    const int* batch_l = (const int*)d_in[6];
    const int* batch_s = (const int*)d_in[7];
    const float* Wa[4] = {(const float*)d_in[8],  (const float*)d_in[12], (const float*)d_in[16], (const float*)d_in[20]};
    const float* ba[4] = {(const float*)d_in[9],  (const float*)d_in[13], (const float*)d_in[17], (const float*)d_in[21]};
    const float* Wb[4] = {(const float*)d_in[10], (const float*)d_in[14], (const float*)d_in[18], (const float*)d_in[22]};
    const float* bb[4] = {(const float*)d_in[11], (const float*)d_in[15], (const float*)d_in[19], (const float*)d_in[23]};
    const float* lin1_W = (const float*)d_in[24];
    const float* lin1_b = (const float*)d_in[25];
    const float* lin2_W = (const float*)d_in[26];
    const float* lin2_b = (const float*)d_in[27];
    const float* gamma  = (const float*)d_in[28];
    const float* beta   = (const float*)d_in[29];
    const float* out_W  = (const float*)d_in[30];
    const float* out_b  = (const float*)d_in[31];

    const int Nl = in_sizes[0] / 64;
    const int El = in_sizes[2];
    const int Ns = in_sizes[3] / 32;
    const int Es = in_sizes[5];
    const int H = 128, G = 512, L2 = 64;
    const int* src_l = ei_l; const int* dst_l = ei_l + El;
    const int* src_s = ei_s; const int* dst_s = ei_s + Es;

    // workspace layout (float region first, then bf16 staging; all 16B aligned)
    float* ws = (float*)d_ws;
    size_t capN = (size_t)Nl * H;
    float* F0     = ws;
    float* F1     = F0 + capN;
    float* XW     = F1 + capN;
    float* dinv_l = XW + capN;
    float* dinv_s = dinv_l + Nl;
    float* norm_l = dinv_s + Ns;
    float* norm_s = norm_l + El;
    float* pool   = norm_s + Es;
    float* cnt    = pool + (size_t)G * H;
    float* t1     = cnt + G;
    float* t2     = t1 + (size_t)G * H;
    Scratch sc;
    sc.Xhi = (__bf16*)(t2 + (size_t)G * L2);
    sc.Xlo = sc.Xhi + capN;
    sc.Wph = sc.Xlo + capN;
    sc.Wpl = sc.Wph + (size_t)H * H;
    (void)ws_size; (void)n_in; (void)out_size;

    // layer-invariant degree / normalization terms (once per branch)
    fill_k<<<cdiv(Nl, 256), 256, 0, stream>>>(dinv_l, 1.0f, Nl);
    deg_accum_k<<<cdiv(El, 256), 256, 0, stream>>>(dst_l, w_l, dinv_l, El);
    rsqrt_k<<<cdiv(Nl, 256), 256, 0, stream>>>(dinv_l, Nl);
    edge_norm_k<<<cdiv(El, 256), 256, 0, stream>>>(src_l, dst_l, w_l, dinv_l, norm_l, El);

    fill_k<<<cdiv(Ns, 256), 256, 0, stream>>>(dinv_s, 1.0f, Ns);
    deg_accum_k<<<cdiv(Es, 256), 256, 0, stream>>>(dst_s, w_s, dinv_s, Es);
    rsqrt_k<<<cdiv(Ns, 256), 256, 0, stream>>>(dinv_s, Ns);
    edge_norm_k<<<cdiv(Es, 256), 256, 0, stream>>>(src_s, dst_s, w_s, dinv_s, norm_s, Es);

    fill_k<<<cdiv((long long)G * H, 256), 256, 0, stream>>>(pool, 0.0f, G * H);
    fill_k<<<cdiv(G, 256), 256, 0, stream>>>(cnt, 0.0f, G);

    // large branch: x_l -> F0 -> F1 -> F0 -> F1
    {
        const float* hin = x_l; int Fin = 64;
        float* bufs[2] = {F0, F1};
        for (int i = 0; i < 4; ++i) {
            float* hout = bufs[i & 1];
            gcn_layer(hin, Fin, Wa[i], ba[i], src_l, dst_l, norm_l, dinv_l, XW, hout, Nl, El, sc, stream);
            hin = hout; Fin = H;
        }
        pool_accum_k<<<cdiv((long long)Nl * 32, 256), 256, 0, stream>>>(F1, batch_l, pool, cnt, Nl, H);
    }
    // small branch (reuses the same buffers)
    {
        const float* hin = x_s; int Fin = 32;
        float* bufs[2] = {F0, F1};
        for (int i = 0; i < 4; ++i) {
            float* hout = bufs[i & 1];
            gcn_layer(hin, Fin, Wb[i], bb[i], src_s, dst_s, norm_s, dinv_s, XW, hout, Ns, Es, sc, stream);
            hin = hout; Fin = H;
        }
        pool_accum_k<<<cdiv((long long)Ns * 32, 256), 256, 0, stream>>>(F1, batch_s, pool, cnt, Ns, H);
    }
    pool_div_k<<<cdiv((long long)G * H, 256), 256, 0, stream>>>(pool, cnt, G, H);

    // MLP head: WMMA GEMMs + BN/relu + final dot
    gemm_full(pool, lin1_W, lin1_b, t1, nullptr, nullptr, G, H, H, 0, sc, stream);
    gemm_full(t1,   lin2_W, lin2_b, t2, nullptr, nullptr, G, H, L2, 0, sc, stream);

    float* outp = (float*)d_out;     // [G] scores first
    float* hO   = outp + G;          // then h [G, 64]
    bn_relu_k<<<cdiv((long long)G * L2, 256), 256, 0, stream>>>(t2, gamma, beta, hO, G, L2);
    out_head_k<<<cdiv(G, 256), 256, 0, stream>>>(hO, out_W, out_b, outp, G, L2);
}
